// HybridCompressedLinear_30571577213783
// MI455X (gfx1250) — matrix operand order repaired and checked by reading, not compiled, not run on hardware
//
#include <hip/hip_runtime.h>
#include <stdint.h>

// ---------------------------------------------------------------------------
// HybridCompressedLinear: y = x @ (W*mask).T + bias as one bf16 WMMA GEMM
//   M = 16384 (B*S), N = 6144 (3*D concat), K = 2048, f32 accumulate.
// Pack x and masked weights to bf16 once; GEMM stages tiles with CDNA5
// GLOBAL_LOAD_ASYNC_TO_LDS (ASYNCcnt) and computes with v_wmma_f32_16x16x32_bf16.
// ---------------------------------------------------------------------------

typedef __attribute__((ext_vector_type(16))) __bf16 v16bf;
typedef __attribute__((ext_vector_type(8)))  float  v8f;
typedef __attribute__((ext_vector_type(4)))  int    i4v;
typedef __attribute__((ext_vector_type(4)))  float  f4v;

#define M_TOTAL 16384
#define N_TOTAL 6144
#define K_TOTAL 2048
#define MT 128          // workgroup M tile
#define NT 256          // workgroup N tile
#define KSTEP 32        // one bf16 WMMA K per stage
#define LDT 40          // padded LDS row pitch in bf16 elements (32 data + 8 pad)
#define NKSTEPS (K_TOTAL / KSTEP)   // 64

// ---- pack kernels ----------------------------------------------------------

__global__ __launch_bounds__(256)
void pack_x_kernel(const float* __restrict__ src, __bf16* __restrict__ dst, int n4) {
    int i = blockIdx.x * blockDim.x + threadIdx.x;
    if (i < n4) {
        f4v v = ((const f4v*)src)[i];
        __bf16* d = dst + (size_t)i * 4;
        d[0] = (__bf16)v.x; d[1] = (__bf16)v.y; d[2] = (__bf16)v.z; d[3] = (__bf16)v.w;
    }
}

__global__ __launch_bounds__(256)
void pack_w_kernel(const float* __restrict__ w, const uint8_t* __restrict__ mask,
                   __bf16* __restrict__ dst, int n4) {
    int i = blockIdx.x * blockDim.x + threadIdx.x;
    if (i < n4) {
        f4v v = ((const f4v*)w)[i];
        const uint8_t* m = mask + (size_t)i * 4;
        __bf16* d = dst + (size_t)i * 4;
        d[0] = m[0] ? (__bf16)v.x : (__bf16)0.0f;
        d[1] = m[1] ? (__bf16)v.y : (__bf16)0.0f;
        d[2] = m[2] ? (__bf16)v.z : (__bf16)0.0f;
        d[3] = m[3] ? (__bf16)v.w : (__bf16)0.0f;
    }
}

// ---- CDNA5 async global->LDS (ISA 15.18.3 op 98, ASYNCcnt) -----------------

__device__ __forceinline__ uint32_t lds_byte_off(const void* p) {
    // LDS aperture occupies bits [63:32] of a flat shared address; low dword
    // is the in-workgroup LDS byte offset (= what VDST of async-to-LDS wants).
    return (uint32_t)(uintptr_t)p;
}

__device__ __forceinline__ void async_ld_b128(uint32_t lds_off, const void* gaddr) {
    asm volatile("global_load_async_to_lds_b128 %0, %1, off"
                 :: "v"(lds_off), "v"(gaddr)
                 : "memory");
}

__device__ __forceinline__ void wait_async0() {
    asm volatile("s_wait_asynccnt 0x0" ::: "memory");
}

// ---- fragment loaders (ISA 7.12.2 layouts) ---------------------------------

// A 16x32 bf16: lanes 0-15 hold row m, K = {0..7, 16..23}; lanes 16-31: K = {8..15, 24..31}
__device__ __forceinline__ v16bf load_frag_a(const __bf16* lds, int m0, int lane) {
    int m  = m0 + (lane & 15);
    int ko = (lane >> 4) * 8;
    const __bf16* p = lds + m * LDT + ko;
    union { i4v i[2]; v16bf v; } u;
    u.i[0] = *(const i4v*)p;          // K = ko .. ko+7
    u.i[1] = *(const i4v*)(p + 16);   // K = ko+16 .. ko+23
    return u.v;
}

// B 32x16 bf16 (stored as W rows, K contiguous): lanes 0-15 hold col n, K = {0..7, 16..23};
// lanes 16-31 hold col n, K = {8..15, 24..31}.
__device__ __forceinline__ v16bf load_frag_b(const __bf16* lds, int n0, int lane) {
    int n  = n0 + (lane & 15);
    int ko = (lane >> 4) * 8;
    const __bf16* p = lds + n * LDT + ko;
    union { i4v i[2]; v16bf v; } u;
    u.i[0] = *(const i4v*)p;
    u.i[1] = *(const i4v*)(p + 16);
    return u.v;
}

// ---- GEMM kernel -----------------------------------------------------------

__global__ __launch_bounds__(256)
void gemm_bf16_wmma_kernel(const __bf16* __restrict__ xbf,     // [M_TOTAL, K_TOTAL]
                           const __bf16* __restrict__ wall,    // [N_TOTAL, K_TOTAL]
                           const float*  __restrict__ bias,    // [N_TOTAL]
                           float*        __restrict__ out)     // [M_TOTAL, N_TOTAL]
{
    __shared__ __bf16 ldsA[2][MT * LDT];   // 2 x 10 KB
    __shared__ __bf16 ldsB[2][NT * LDT];   // 2 x 20 KB

    const int t    = threadIdx.x;
    const int lane = t & 31;
    const int wave = t >> 5;          // 8 waves
    const int wm   = wave >> 2;       // 0..1 : wave M position (64 rows)
    const int wn   = wave & 3;        // 0..3 : wave N position (64 cols)

    const int mbase = blockIdx.y * MT;
    const int nbase = blockIdx.x * NT;

    // ---- async staging assignment ----
    // A tile: 128 rows x 32 bf16 (64B/row). 256 threads -> 2 threads/row, 32B each.
    const int rowA = t >> 1;
    const int segA = t & 1;                      // which 32B half of the row
    const __bf16* gA = xbf + (size_t)(mbase + rowA) * K_TOTAL + segA * 16;
    const uint32_t aOff[2] = {
        lds_byte_off(&ldsA[0][rowA * LDT + segA * 16]),
        lds_byte_off(&ldsA[1][rowA * LDT + segA * 16])
    };
    // B tile: 256 rows x 32 bf16 (64B/row). 1 thread/row, 64B each.
    const __bf16* gB = wall + (size_t)(nbase + t) * K_TOTAL;
    const uint32_t bOff[2] = {
        lds_byte_off(&ldsB[0][t * LDT]),
        lds_byte_off(&ldsB[1][t * LDT])
    };

    v8f acc[4][4];
#pragma unroll
    for (int mf = 0; mf < 4; ++mf)
#pragma unroll
        for (int nf = 0; nf < 4; ++nf)
            acc[mf][nf] = (v8f){0.f, 0.f, 0.f, 0.f, 0.f, 0.f, 0.f, 0.f};

    // ---- prologue: async-stage K-step 0 into buffer 0 ----
    {
        const __bf16* pA = gA;
        async_ld_b128(aOff[0],      pA);
        async_ld_b128(aOff[0] + 16, pA + 8);
        const __bf16* pB = gB;
        async_ld_b128(bOff[0],      pB);
        async_ld_b128(bOff[0] + 16, pB + 8);
        async_ld_b128(bOff[0] + 32, pB + 16);
        async_ld_b128(bOff[0] + 48, pB + 24);
    }
    wait_async0();
    __syncthreads();

    for (int ks = 0; ks < NKSTEPS; ++ks) {
        const int cur = ks & 1;
        const int nxt = cur ^ 1;
        const bool has_next = (ks + 1) < NKSTEPS;

        // issue next-stage async copies while we compute on `cur`
        if (has_next) {
            const __bf16* pA = gA + (ks + 1) * KSTEP;
            async_ld_b128(aOff[nxt],      pA);
            async_ld_b128(aOff[nxt] + 16, pA + 8);
            const __bf16* pB = gB + (ks + 1) * KSTEP;
            async_ld_b128(bOff[nxt],      pB);
            async_ld_b128(bOff[nxt] + 16, pB + 8);
            async_ld_b128(bOff[nxt] + 32, pB + 16);
            async_ld_b128(bOff[nxt] + 48, pB + 24);
        }

        // fragments from current LDS buffer
        v16bf afrag[4], bfrag[4];
#pragma unroll
        for (int mf = 0; mf < 4; ++mf)
            afrag[mf] = load_frag_a(ldsA[cur], wm * 64 + mf * 16, lane);
#pragma unroll
        for (int nf = 0; nf < 4; ++nf)
            bfrag[nf] = load_frag_b(ldsB[cur], wn * 64 + nf * 16, lane);

        // 16 independent WMMAs per wave per K-step
#pragma unroll
        for (int mf = 0; mf < 4; ++mf)
#pragma unroll
            for (int nf = 0; nf < 4; ++nf)
                acc[mf][nf] = __builtin_amdgcn_wmma_f32_16x16x32_bf16(
                    /*neg_a=*/false, afrag[mf],
                    /*neg_b=*/false, bfrag[nf],
                    /*c_mod=*/(short)0, acc[mf][nf],
                    /*reuse_a=*/false, /*reuse_b=*/false);

        if (has_next) wait_async0();
        __syncthreads();
    }

    // ---- epilogue: C layout -> VGPR r holds M = r + 8*(lane>=16), N = lane&15
    const int col  = lane & 15;
    const int rsel = lane >> 4;
#pragma unroll
    for (int nf = 0; nf < 4; ++nf) {
        const int n = nbase + wn * 64 + nf * 16 + col;
        const float bv = bias[n];
#pragma unroll
        for (int mf = 0; mf < 4; ++mf) {
            const int mrow = mbase + wm * 64 + mf * 16 + rsel * 8;
#pragma unroll
            for (int r = 0; r < 8; ++r) {
                out[(size_t)(mrow + r) * N_TOTAL + n] = acc[mf][nf][r] + bv;
            }
        }
    }
}

// ---- launch ----------------------------------------------------------------

extern "C" void kernel_launch(void* const* d_in, const int* in_sizes, int n_in,
                              void* d_out, int out_size, void* d_ws, size_t ws_size,
                              hipStream_t stream) {
    const float*   x    = (const float*)d_in[0];
    const float*   w_q  = (const float*)d_in[1];
    const float*   w_k  = (const float*)d_in[2];
    const float*   w_v  = (const float*)d_in[3];
    const uint8_t* m_q  = (const uint8_t*)d_in[4];
    const uint8_t* m_k  = (const uint8_t*)d_in[5];
    const uint8_t* m_v  = (const uint8_t*)d_in[6];
    const float*   bias = (const float*)d_in[7];
    float* out = (float*)d_out;

    // workspace layout: xbf [16384*2048] bf16 | wall [6144*2048] bf16  (~92 MB)
    __bf16* xbf  = (__bf16*)d_ws;
    __bf16* wall = xbf + (size_t)M_TOTAL * K_TOTAL;

    const int xn4 = (M_TOTAL * K_TOTAL) / 4;
    const int wn4 = (K_TOTAL * K_TOTAL) / 4;

    pack_x_kernel<<<(xn4 + 255) / 256, 256, 0, stream>>>(x, xbf, xn4);
    pack_w_kernel<<<(wn4 + 255) / 256, 256, 0, stream>>>(w_q, m_q, wall,                                 wn4);
    pack_w_kernel<<<(wn4 + 255) / 256, 256, 0, stream>>>(w_k, m_k, wall + (size_t)K_TOTAL * K_TOTAL,     wn4);
    pack_w_kernel<<<(wn4 + 255) / 256, 256, 0, stream>>>(w_v, m_v, wall + (size_t)2 * K_TOTAL * K_TOTAL, wn4);

    dim3 grid(N_TOTAL / NT, M_TOTAL / MT);   // (24, 128)
    gemm_bf16_wmma_kernel<<<grid, 256, 0, stream>>>(xbf, wall, bias, out);
}
